// Llama4MoE_25245817766057
// MI455X (gfx1250) — matrix operand order repaired and checked by reading, not compiled, not run on hardware
//
#include <hip/hip_runtime.h>
#include <hip/hip_bf16.h>

// Problem sizes (fixed by the reference)
#define E_ 8
#define H_ 2048
#define I_ 4096
#define T_ 2048   // B*S tokens

// GEMM tiling
#define BM 64
#define BN 128
#define BK 32
#define KPAD 40          // bf16 elems per LDS row (80B, 16B aligned, conflict-dodging)
#define NTHREADS 256     // 8 waves (wave32)

typedef __bf16 bf16;
typedef __attribute__((ext_vector_type(16))) __bf16 v16bf;
typedef __attribute__((ext_vector_type(8)))  __bf16 v8bf;
typedef __attribute__((ext_vector_type(8)))  float  v8f;
typedef __attribute__((address_space(3)))    bf16   lds_bf16;

__device__ __forceinline__ float f4c(const float4& v, int i) {
  switch (i) { case 0: return v.x; case 1: return v.y; case 2: return v.z; default: return v.w; }
}

// gfx1250 async global->LDS copy (16B per lane), tracked by ASYNCcnt
__device__ __forceinline__ void async_copy_b128(bf16* lds_dst, const bf16* gsrc) {
  lds_bf16* l = (lds_bf16*)lds_dst;
  asm volatile("global_load_async_to_lds_b128 %0, %1, off"
               :: "v"(l), "v"(gsrc)
               : "memory");
}
__device__ __forceinline__ void wait_async0() {
  asm volatile("s_wait_asynccnt 0x0" ::: "memory");
}

// ---------------------------------------------------------------------------
// 1) Router: one wave per token. logits = x @ router_w^T, top-1, sigmoid.
// ---------------------------------------------------------------------------
__global__ void moe_router(const float* __restrict__ x, const float* __restrict__ rw,
                           int* __restrict__ counts, int* __restrict__ eidx,
                           int* __restrict__ pos, float* __restrict__ score) {
  const int t = blockIdx.x * 4 + (threadIdx.x >> 5);
  const int lane = threadIdx.x & 31;
  float acc[E_];
#pragma unroll
  for (int e = 0; e < E_; ++e) acc[e] = 0.0f;
  const float* xp = x + (size_t)t * H_;
  for (int i = lane; i < H_; i += 32) {
    const float xv = xp[i];
#pragma unroll
    for (int e = 0; e < E_; ++e) acc[e] += xv * rw[e * H_ + i];
  }
#pragma unroll
  for (int off = 16; off > 0; off >>= 1) {
#pragma unroll
    for (int e = 0; e < E_; ++e) acc[e] += __shfl_down(acc[e], off, 32);
  }
  if (lane == 0) {
    int best = 0; float bv = acc[0];
#pragma unroll
    for (int e = 1; e < E_; ++e) { if (acc[e] > bv) { bv = acc[e]; best = e; } }
    const int p = atomicAdd(&counts[best], 1);
    eidx[t] = best;
    pos[t]  = p;
    score[t] = 1.0f / (1.0f + __expf(-bv));
  }
}

// ---------------------------------------------------------------------------
// 2) Exclusive prefix over 8 expert counts -> counts[8..15] = row bases
// ---------------------------------------------------------------------------
__global__ void moe_prefix(int* __restrict__ counts) {
  if (threadIdx.x == 0 && blockIdx.x == 0) {
    int b = 0;
#pragma unroll
    for (int e = 0; e < E_; ++e) { counts[8 + e] = b; b += counts[e]; }
  }
}

// ---------------------------------------------------------------------------
// 3) Scatter: gathered bf16 A (x*score) per expert + unscaled bf16 copy
// ---------------------------------------------------------------------------
__global__ void moe_scatter(const float* __restrict__ x, const int* __restrict__ counts,
                            const int* __restrict__ eidx, const int* __restrict__ pos,
                            const float* __restrict__ score, int* __restrict__ rowmap,
                            bf16* __restrict__ Art, bf16* __restrict__ Ash) {
  const int t = blockIdx.x;
  const int tid = threadIdx.x;
  const int e = eidx[t];
  const int r = counts[8 + e] + pos[t];
  const float s = score[t];
  if (tid == 0) rowmap[r] = t;
  const float4* xp = (const float4*)(x + (size_t)t * H_);
  bf16* sh = Ash + (size_t)t * H_;
  bf16* rt = Art + (size_t)r * H_;
  for (int i = tid; i < H_ / 4; i += 256) {
    const float4 v = xp[i];
    union { bf16 h[4]; uint2 u; } a, b;
    a.h[0] = (bf16)v.x; a.h[1] = (bf16)v.y; a.h[2] = (bf16)v.z; a.h[3] = (bf16)v.w;
    b.h[0] = (bf16)(v.x * s); b.h[1] = (bf16)(v.y * s);
    b.h[2] = (bf16)(v.z * s); b.h[3] = (bf16)(v.w * s);
    *(uint2*)(sh + i * 4) = a.u;
    *(uint2*)(rt + i * 4) = b.u;
  }
}

// ---------------------------------------------------------------------------
// Fragment helpers (ISA 7.12.2 layouts)
// A 16x32 bf16: lane half h, elems j<8 -> K=8h+j ; j>=8 -> K=16+8h+(j-8)
// B 32x16 bf16 stored [n][k] in LDS: elems j<8 -> K=8h+j ; j>=8 -> K=16+8h+(j-8) at fixed N=lane&15
// ---------------------------------------------------------------------------
__device__ __forceinline__ v16bf frag_a(const bf16* As, int m, int half) {
  const v8bf lo = *(const v8bf*)(&As[m * KPAD + half * 8]);
  const v8bf hi = *(const v8bf*)(&As[m * KPAD + 16 + half * 8]);
  return __builtin_shufflevector(lo, hi, 0,1,2,3,4,5,6,7,8,9,10,11,12,13,14,15);
}
__device__ __forceinline__ v16bf frag_b(const bf16* Bs, int n, int half) {
  const v8bf lo = *(const v8bf*)(&Bs[n * KPAD + half * 8]);
  const v8bf hi = *(const v8bf*)(&Bs[n * KPAD + 16 + half * 8]);
  return __builtin_shufflevector(lo, hi, 0,1,2,3,4,5,6,7,8,9,10,11,12,13,14,15);
}

__device__ __forceinline__ void pack4(bf16* dst, const float4& a, const float4& b,
                                      const float4& c, const float4& d, int n) {
  union { bf16 h[4]; uint2 u; } p;
  p.h[0] = (bf16)f4c(a, n); p.h[1] = (bf16)f4c(b, n);
  p.h[2] = (bf16)f4c(c, n); p.h[3] = (bf16)f4c(d, n);
  *(uint2*)dst = p.u;
}

// ---------------------------------------------------------------------------
// 4) GEMM1: H = silu(A@Bg) * (A@Bu). Double-buffered LDS, async A DMA.
// ---------------------------------------------------------------------------
template <bool ROUTED>
__global__ __launch_bounds__(NTHREADS)
void moe_gemm1(const bf16* __restrict__ Abase, const int* __restrict__ counts,
               const float* __restrict__ BgBase, const float* __restrict__ BuBase,
               bf16* __restrict__ Hbase) {
  __shared__ bf16 As[2][BM * KPAD];
  __shared__ bf16 Bgs[2][BN * KPAD];
  __shared__ bf16 Bus[2][BN * KPAD];

  int nrows;
  const bf16* A; const float* Bg; const float* Bu; size_t ldB; bf16* Hout;
  if (ROUTED) {
    const int e = blockIdx.z;
    nrows = counts[e];
    if ((int)blockIdx.y * BM >= nrows) return;   // block-uniform: EXEC stays full
    const int base = counts[8 + e];
    A   = Abase + (size_t)base * H_;
    Bg  = BgBase + (size_t)e * H_ * (2 * I_);
    Bu  = Bg + I_;
    ldB = 2 * I_;
    Hout = Hbase + (size_t)base * I_;
  } else {
    nrows = T_;
    A = Abase; Bg = BgBase; Bu = BuBase; ldB = I_; Hout = Hbase;
  }

  const int tid  = threadIdx.x;
  const int lane = tid & 31;
  const int half = lane >> 4;
  const int ncl  = lane & 15;
  const int w    = tid >> 5;
  const int mwave = (w >> 2) * 32;
  const int nwave = (w & 3) * 32;
  const int ncol0 = blockIdx.x * BN;
  const int row0  = blockIdx.y * BM;

  v8f cg[2][2], cu[2][2];
#pragma unroll
  for (int i = 0; i < 2; ++i)
#pragma unroll
    for (int j = 0; j < 2; ++j) { cg[i][j] = (v8f){}; cu[i][j] = (v8f){}; }

  const int arow = tid >> 2;            // 0..63
  const int akc  = (tid & 3) * 8;       // 0,8,16,24
  const int bn0  = (tid & 31) * 4;      // 0..124
  const int bkl  = (tid >> 5) * 4;      // 0..28
  // clamp instead of predicate: EXEC stays full, no garbage rows in LDS
  const int grow = min(row0 + arow, nrows - 1);
  const bf16* aptr = A + (size_t)grow * H_ + akc;
  const float* bgp = Bg + (size_t)bkl * ldB + ncol0 + bn0;
  const float* bup = Bu + (size_t)bkl * ldB + ncol0 + bn0;

  // ---- software pipeline prologue: tile 0 in flight ----
  async_copy_b128(&As[0][arow * KPAD + akc], aptr);
  float4 g0 = *(const float4*)(bgp + 0 * ldB);
  float4 g1 = *(const float4*)(bgp + 1 * ldB);
  float4 g2 = *(const float4*)(bgp + 2 * ldB);
  float4 g3 = *(const float4*)(bgp + 3 * ldB);
  float4 u0 = *(const float4*)(bup + 0 * ldB);
  float4 u1 = *(const float4*)(bup + 1 * ldB);
  float4 u2 = *(const float4*)(bup + 2 * ldB);
  float4 u3 = *(const float4*)(bup + 3 * ldB);

  for (int k0 = 0, it = 0; k0 < H_; k0 += BK, ++it) {
    const int b = it & 1;
    // stage current B regs into LDS[b] (conflict with compute(it-2) fenced by barrier(it-1))
#pragma unroll
    for (int n = 0; n < 4; ++n) {
      pack4(&Bgs[b][(bn0 + n) * KPAD + bkl], g0, g1, g2, g3, n);
      pack4(&Bus[b][(bn0 + n) * KPAD + bkl], u0, u1, u2, u3, n);
    }
    const bool more = (k0 + BK) < H_;
    if (more) {  // next B tile: in flight across barrier + compute
      const size_t koff = (size_t)(k0 + BK) * ldB;
      g0 = *(const float4*)(bgp + koff + 0 * ldB);
      g1 = *(const float4*)(bgp + koff + 1 * ldB);
      g2 = *(const float4*)(bgp + koff + 2 * ldB);
      g3 = *(const float4*)(bgp + koff + 3 * ldB);
      u0 = *(const float4*)(bup + koff + 0 * ldB);
      u1 = *(const float4*)(bup + koff + 1 * ldB);
      u2 = *(const float4*)(bup + koff + 2 * ldB);
      u3 = *(const float4*)(bup + koff + 3 * ldB);
    }
    wait_async0();                       // this wave's async A(it) landed
    __syncthreads();                     // all tiles of buffer b staged
    if (more)                            // post-barrier: compute(it-1) on b^1 is done
      async_copy_b128(&As[b ^ 1][arow * KPAD + akc], aptr + k0 + BK);

    // --- WMMA compute on buffer b ---
    v16bf af[2];
#pragma unroll
    for (int mt = 0; mt < 2; ++mt) af[mt] = frag_a(As[b], mwave + mt * 16 + ncl, half);
#pragma unroll
    for (int nt = 0; nt < 2; ++nt) {
      const int n = nwave + nt * 16 + ncl;
      const v16bf bG = frag_b(Bgs[b], n, half);
      const v16bf bU = frag_b(Bus[b], n, half);
#pragma unroll
      for (int mt = 0; mt < 2; ++mt) {
        cg[mt][nt] = __builtin_amdgcn_wmma_f32_16x16x32_bf16(false, af[mt], false, bG,
                                                             (short)0, cg[mt][nt], false, false);
        cu[mt][nt] = __builtin_amdgcn_wmma_f32_16x16x32_bf16(false, af[mt], false, bU,
                                                             (short)0, cu[mt][nt], false, false);
      }
    }
  }

  // --- fused SiLU epilogue -> bf16 intermediate ---
  const bool full = !ROUTED || (row0 + BM <= nrows);  // block-uniform fast path
#pragma unroll
  for (int mt = 0; mt < 2; ++mt)
#pragma unroll
    for (int nt = 0; nt < 2; ++nt) {
      const int c = ncol0 + nwave + nt * 16 + ncl;
      bf16* hp = Hout + (size_t)(row0 + mwave + mt * 16 + 8 * half) * I_ + c;
      if (full) {
#pragma unroll
        for (int g = 0; g < 8; ++g) {
          const float gate = cg[mt][nt][g];
          const float up   = cu[mt][nt][g];
          const float sig  = __builtin_amdgcn_rcpf(1.0f + __expf(-gate));
          hp[(size_t)g * I_] = (bf16)(up * gate * sig);
        }
      } else {
#pragma unroll
        for (int g = 0; g < 8; ++g) {
          const int r = row0 + mwave + mt * 16 + 8 * half + g;
          if (r < nrows) {
            const float gate = cg[mt][nt][g];
            const float up   = cu[mt][nt][g];
            const float sig  = __builtin_amdgcn_rcpf(1.0f + __expf(-gate));
            hp[(size_t)g * I_] = (bf16)(up * gate * sig);
          }
        }
      }
    }
}

// ---------------------------------------------------------------------------
// 5) GEMM2: out = H @ Bdown ; shared writes '=', routed scatters '+='
// ---------------------------------------------------------------------------
template <bool ROUTED>
__global__ __launch_bounds__(NTHREADS)
void moe_gemm2(const bf16* __restrict__ Hbase, const int* __restrict__ counts,
               const int* __restrict__ rowmap, const float* __restrict__ Bbase,
               float* __restrict__ out) {
  __shared__ bf16 As[2][BM * KPAD];
  __shared__ bf16 Bs[2][BN * KPAD];

  int nrows; const bf16* A; const float* B; const int* rmap;
  if (ROUTED) {
    const int e = blockIdx.z;
    nrows = counts[e];
    if ((int)blockIdx.y * BM >= nrows) return;
    const int base = counts[8 + e];
    A = Hbase + (size_t)base * I_;
    B = Bbase + (size_t)e * I_ * H_;
    rmap = rowmap + base;
  } else {
    nrows = T_; A = Hbase; B = Bbase; rmap = nullptr;
  }
  const size_t ldB = H_;

  const int tid  = threadIdx.x;
  const int lane = tid & 31;
  const int half = lane >> 4;
  const int ncl  = lane & 15;
  const int w    = tid >> 5;
  const int mwave = (w >> 2) * 32;
  const int nwave = (w & 3) * 32;
  const int ncol0 = blockIdx.x * BN;
  const int row0  = blockIdx.y * BM;

  v8f cc[2][2];
#pragma unroll
  for (int i = 0; i < 2; ++i)
#pragma unroll
    for (int j = 0; j < 2; ++j) cc[i][j] = (v8f){};

  const int arow = tid >> 2;
  const int akc  = (tid & 3) * 8;
  const int bn0  = (tid & 31) * 4;
  const int bkl  = (tid >> 5) * 4;
  const int grow = min(row0 + arow, nrows - 1);
  const bf16* aptr = A + (size_t)grow * I_ + akc;
  const float* bp = B + (size_t)bkl * ldB + ncol0 + bn0;

  async_copy_b128(&As[0][arow * KPAD + akc], aptr);
  float4 b0 = *(const float4*)(bp + 0 * ldB);
  float4 b1 = *(const float4*)(bp + 1 * ldB);
  float4 b2 = *(const float4*)(bp + 2 * ldB);
  float4 b3 = *(const float4*)(bp + 3 * ldB);

  for (int k0 = 0, it = 0; k0 < I_; k0 += BK, ++it) {
    const int b = it & 1;
#pragma unroll
    for (int n = 0; n < 4; ++n)
      pack4(&Bs[b][(bn0 + n) * KPAD + bkl], b0, b1, b2, b3, n);
    const bool more = (k0 + BK) < I_;
    if (more) {
      const size_t koff = (size_t)(k0 + BK) * ldB;
      b0 = *(const float4*)(bp + koff + 0 * ldB);
      b1 = *(const float4*)(bp + koff + 1 * ldB);
      b2 = *(const float4*)(bp + koff + 2 * ldB);
      b3 = *(const float4*)(bp + koff + 3 * ldB);
    }
    wait_async0();
    __syncthreads();
    if (more)
      async_copy_b128(&As[b ^ 1][arow * KPAD + akc], aptr + k0 + BK);

    v16bf af[2];
#pragma unroll
    for (int mt = 0; mt < 2; ++mt) af[mt] = frag_a(As[b], mwave + mt * 16 + ncl, half);
#pragma unroll
    for (int nt = 0; nt < 2; ++nt) {
      const v16bf bF = frag_b(Bs[b], nwave + nt * 16 + ncl, half);
#pragma unroll
      for (int mt = 0; mt < 2; ++mt)
        cc[mt][nt] = __builtin_amdgcn_wmma_f32_16x16x32_bf16(false, af[mt], false, bF,
                                                             (short)0, cc[mt][nt], false, false);
    }
  }

  const bool full = !ROUTED || (row0 + BM <= nrows);
#pragma unroll
  for (int mt = 0; mt < 2; ++mt)
#pragma unroll
    for (int nt = 0; nt < 2; ++nt) {
      const int c = ncol0 + nwave + nt * 16 + ncl;
#pragma unroll
      for (int g = 0; g < 8; ++g) {
        const int r = row0 + mwave + mt * 16 + 8 * half + g;
        if (full || r < nrows) {
          const int trow = ROUTED ? rmap[r] : r;
          float* p = out + (size_t)trow * H_ + c;
          if (ROUTED) *p += cc[mt][nt][g];   // top-1: unique writer per (token,col)
          else        *p  = cc[mt][nt][g];
        }
      }
    }
}

// ---------------------------------------------------------------------------
// Host launch (graph-capture safe: only async ops on `stream`)
// ---------------------------------------------------------------------------
extern "C" void kernel_launch(void* const* d_in, const int* in_sizes, int n_in,
                              void* d_out, int out_size, void* d_ws, size_t ws_size,
                              hipStream_t stream) {
  const float* x  = (const float*)d_in[0];
  const float* rw = (const float*)d_in[1];
  const float* gu = (const float*)d_in[2];
  const float* dn = (const float*)d_in[3];
  const float* sg = (const float*)d_in[4];
  const float* su = (const float*)d_in[5];
  const float* sd = (const float*)d_in[6];
  float* out = (float*)d_out;

  char* p = (char*)d_ws;
  auto alloc = [&](size_t bytes) -> char* {
    char* r = p; p += (bytes + 255) & ~(size_t)255; return r;
  };
  int*   counts = (int*)  alloc(16 * sizeof(int));          // [0..7]=counts, [8..15]=bases
  int*   eidx   = (int*)  alloc((size_t)T_ * sizeof(int));
  int*   pos    = (int*)  alloc((size_t)T_ * sizeof(int));
  float* score  = (float*)alloc((size_t)T_ * sizeof(float));
  int*   rowmap = (int*)  alloc((size_t)T_ * sizeof(int));
  bf16*  Ash    = (bf16*) alloc((size_t)T_ * H_ * sizeof(bf16));
  bf16*  Art    = (bf16*) alloc((size_t)T_ * H_ * sizeof(bf16));
  bf16*  Hsh    = (bf16*) alloc((size_t)T_ * I_ * sizeof(bf16));
  bf16*  Hrt    = (bf16*) alloc((size_t)T_ * I_ * sizeof(bf16));
  (void)in_sizes; (void)n_in; (void)out_size; (void)ws_size;

  hipMemsetAsync(counts, 0, 16 * sizeof(int), stream);

  moe_router <<<T_ / 4, 128, 0, stream>>>(x, rw, counts, eidx, pos, score);
  moe_prefix <<<1, 32, 0, stream>>>(counts);
  moe_scatter<<<T_, 256, 0, stream>>>(x, counts, eidx, pos, score, rowmap, Art, Ash);

  // shared expert: SwiGLU up-projection, then routed experts (gathered rows)
  moe_gemm1<false><<<dim3(I_ / BN, T_ / BM, 1),  NTHREADS, 0, stream>>>(Ash, counts, sg, su, Hsh);
  moe_gemm1<true> <<<dim3(I_ / BN, T_ / BM, E_), NTHREADS, 0, stream>>>(Art, counts, gu, nullptr, Hrt);

  // down-projections: shared writes '=', routed adds '+=' (stream-ordered)
  moe_gemm2<false><<<dim3(H_ / BN, T_ / BM, 1),  NTHREADS, 0, stream>>>(Hsh, counts, nullptr, sd, out);
  moe_gemm2<true> <<<dim3(H_ / BN, T_ / BM, E_), NTHREADS, 0, stream>>>(Hrt, counts, rowmap, dn, out);
}
